// DotProductAttention_47854525612275
// MI455X (gfx1250) — matrix operand order repaired and checked by reading, not compiled, not run on hardware
//
#include <hip/hip_runtime.h>

// ---------------------------------------------------------------------------
// Causal dot-product attention fwd for B=16, S=2048, D=64 (fp32 in/out),
// returning (attention_output, attn_weights).  MI455X / gfx1250, wave32.
//
// One wave32 owns one 16-row query tile.  Two-pass flash softmax:
//   pass 1: m,l per row (online, exact)   -- 4 x v_wmma_f32_16x16x32_f16 / 32 keys
//   pass 2: P = exp(s-m)/l  -> NT-store to attn_weights, LDS relayout C->A,
//           O += P*V                      -- 8 x v_wmma per 32 keys
// Upper triangle of attn_weights is zero-filled with NT b128 stores.
// ---------------------------------------------------------------------------

typedef __attribute__((ext_vector_type(16))) _Float16 v16h;
typedef __attribute__((ext_vector_type(8)))  float    v8f;
typedef __attribute__((ext_vector_type(4)))  float    v4f;   // native vec for NT b128

namespace {

constexpr int BB = 16;
constexpr int SS = 2048;
constexpr int DD = 64;
constexpr float SCALE = 0.125f;   // 1/sqrt(64)
constexpr float NEGI  = -1e9f;    // reference LookAheadMask value

union HF {
  v16h     v;
  _Float16 h[16];
  uint4    q[2];
};

__device__ __forceinline__ v8f wmma16(v16h a, v16h b, v8f c) {
  // D = A(16x32 f16) * B(32x16 f16) + C(16x16 f32)
  return __builtin_amdgcn_wmma_f32_16x16x32_f16(
      /*neg_a=*/false, a, /*neg_b=*/false, b,
      /*c_mod=*/(short)0, c, /*reuse_a=*/false, /*reuse_b=*/false);
}

// reductions across the 16 lanes of each wave32 half (masks < 16 stay in-half)
__device__ __forceinline__ float redmax16(float v) {
#pragma unroll
  for (int m = 1; m <= 8; m <<= 1) v = fmaxf(v, __shfl_xor(v, m, 32));
  return v;
}
__device__ __forceinline__ float redsum16(float v) {
#pragma unroll
  for (int m = 1; m <= 8; m <<= 1) v += __shfl_xor(v, m, 32);
  return v;
}

} // namespace

__global__ __launch_bounds__(256)
void fa_causal_fwd(const float* __restrict__ Q, const float* __restrict__ K,
                   const float* __restrict__ V, float* __restrict__ O,
                   float* __restrict__ W) {
  // per-wave P relayout slab: 16 rows x 32 cols f16, row stride 40 halves
  // (80B: 16B aligned, distinct starting banks across rows)
  __shared__ __align__(16) _Float16 ldsP[8][16 * 40];

  const int tid  = threadIdx.x;
  const int wv   = tid >> 5;
  const int lane = tid & 31;
  const int r    = lane & 15;   // row (A/C) or column (B/C) index
  const int hi   = lane >> 4;   // lane half selects K-range / M-offset

  const int gw    = blockIdx.x * 8 + wv;   // 2048 waves total
  const int b     = gw >> 7;
  const int qt    = gw & 127;
  const int qbase = qt << 4;

  const float* Qb = Q + ((size_t)b * SS + qbase) * DD;
  const float* Kb = K + (size_t)b * SS * DD;
  const float* Vb = V + (size_t)b * SS * DD;
  float*       Ob = O + ((size_t)b * SS + qbase) * DD;
  float*       Wb = W + ((size_t)b * SS + qbase) * (size_t)SS;

  // ---- Q A-fragments (d 0..31 and 32..63), contiguous k = 16*hi + s layout ------
  const float* qrow = Qb + (size_t)r * DD;
  HF aq0, aq1;
#pragma unroll
  for (int i = 0; i < 8; ++i) {
    aq0.h[i]     = (_Float16)qrow[hi * 16 + i];
    aq0.h[i + 8] = (_Float16)qrow[hi * 16 + 8 + i];
    aq1.h[i]     = (_Float16)qrow[32 + hi * 16 + i];
    aq1.h[i + 8] = (_Float16)qrow[32 + hi * 16 + 8 + i];
  }

  float m[8], l[8];
#pragma unroll
  for (int g = 0; g < 8; ++g) { m[g] = -3.0e38f; l[g] = 0.f; }

  const int nblk = (qbase + 15) / 32 + 1;   // causal 32-key blocks to compute
  const int kend = nblk * 32;

  // ======================= pass 1: row max / sumexp ========================
  for (int kb = 0; kb < kend; kb += 32) {
    HF bk00, bk01, bk10, bk11;  // K^T B-frags: keys {kb,kb+16} x d {0..31,32..63}
    const float* k0 = Kb + (size_t)(kb + r) * DD + hi * 16;
    const float* k1 = Kb + (size_t)(kb + 16 + r) * DD + hi * 16;
#pragma unroll
    for (int i = 0; i < 16; ++i) {
      bk00.h[i] = (_Float16)k0[i];
      bk01.h[i] = (_Float16)k0[32 + i];
      bk10.h[i] = (_Float16)k1[i];
      bk11.h[i] = (_Float16)k1[32 + i];
    }
    v8f c0 = {0.f, 0.f, 0.f, 0.f, 0.f, 0.f, 0.f, 0.f};
    v8f c1 = {0.f, 0.f, 0.f, 0.f, 0.f, 0.f, 0.f, 0.f};
    c0 = wmma16(aq0.v, bk00.v, c0);
    c0 = wmma16(aq1.v, bk01.v, c0);
    c1 = wmma16(aq0.v, bk10.v, c1);
    c1 = wmma16(aq1.v, bk11.v, c1);

#pragma unroll
    for (int g = 0; g < 8; ++g) {
      const int rowq = qbase + g + 8 * hi;
      float t0 = (kb + r      <= rowq) ? c0[g] * SCALE : NEGI;
      float t1 = (kb + 16 + r <= rowq) ? c1[g] * SCALE : NEGI;
      float mx = redmax16(fmaxf(t0, t1));
      float mn = fmaxf(m[g], mx);
      float sm = redsum16(__expf(t0 - mn) + __expf(t1 - mn));
      l[g] = l[g] * __expf(m[g] - mn) + sm;
      m[g] = mn;
    }
  }

  float rl[8];
#pragma unroll
  for (int g = 0; g < 8; ++g) rl[g] = 1.0f / l[g];

  v8f oacc[4];
#pragma unroll
  for (int db = 0; db < 4; ++db)
    oacc[db] = (v8f){0.f, 0.f, 0.f, 0.f, 0.f, 0.f, 0.f, 0.f};

  // ============== pass 2: weights out + O += P*V ===========================
  for (int kb = 0; kb < kend; kb += 32) {
    HF bk00, bk01, bk10, bk11;
    const float* k0 = Kb + (size_t)(kb + r) * DD + hi * 16;
    const float* k1 = Kb + (size_t)(kb + 16 + r) * DD + hi * 16;
#pragma unroll
    for (int i = 0; i < 16; ++i) {
      bk00.h[i] = (_Float16)k0[i];
      bk01.h[i] = (_Float16)k0[32 + i];
      bk10.h[i] = (_Float16)k1[i];
      bk11.h[i] = (_Float16)k1[32 + i];
    }
    v8f c0 = {0.f, 0.f, 0.f, 0.f, 0.f, 0.f, 0.f, 0.f};
    v8f c1 = {0.f, 0.f, 0.f, 0.f, 0.f, 0.f, 0.f, 0.f};
    c0 = wmma16(aq0.v, bk00.v, c0);
    c0 = wmma16(aq1.v, bk01.v, c0);
    c1 = wmma16(aq0.v, bk10.v, c1);
    c1 = wmma16(aq1.v, bk11.v, c1);

#pragma unroll
    for (int g = 0; g < 8; ++g) {
      const int rowq = qbase + g + 8 * hi;
      float t0 = (kb + r      <= rowq) ? c0[g] * SCALE : NEGI;
      float t1 = (kb + 16 + r <= rowq) ? c1[g] * SCALE : NEGI;
      float p0 = __expf(t0 - m[g]) * rl[g];   // exact 0 when masked
      float p1 = __expf(t1 - m[g]) * rl[g];
      float* wr = Wb + (size_t)(g + 8 * hi) * SS + kb;
      __builtin_nontemporal_store(p0, wr + r);        // streaming: never re-read
      __builtin_nontemporal_store(p1, wr + 16 + r);
      ldsP[wv][(g + 8 * hi) * 40 + r]      = (_Float16)p0;
      ldsP[wv][(g + 8 * hi) * 40 + 16 + r] = (_Float16)p1;
    }

    // C/D layout -> A layout via per-wave LDS.  DS ops are in-order within a
    // wave, so a compiler fence + dscnt wait is sufficient (no __syncthreads:
    // waves have divergent trip counts).
    asm volatile("s_wait_dscnt 0" ::: "memory");
    HF ap;
    ap.q[0] = *(const uint4*)&ldsP[wv][r * 40 + hi * 16];
    ap.q[1] = *(const uint4*)&ldsP[wv][r * 40 + hi * 16 + 8];

#pragma unroll
    for (int db = 0; db < 4; ++db) {
      HF bv;  // V B-frag: k = key in-block, n = d column (contiguous k = 16*hi + i)
      const float* vb = Vb + (size_t)(kb + hi * 16) * DD + db * 16 + r;
#pragma unroll
      for (int i = 0; i < 16; ++i) bv.h[i] = (_Float16)vb[(size_t)i * DD];
      oacc[db] = wmma16(ap.v, bv.v, oacc[db]);
    }
  }

  // ---- zero-fill strictly-future attention weights (NT b128 streaming) ----
  const v4f z4 = {0.f, 0.f, 0.f, 0.f};
#pragma unroll
  for (int g = 0; g < 8; ++g) {
    v4f* wr4 = (v4f*)(Wb + (size_t)(g + 8 * hi) * SS);
    for (int c4 = kend / 4 + r; c4 < SS / 4; c4 += 16)
      __builtin_nontemporal_store(z4, wr4 + c4);
  }

  // ---- write O (rows g+8*hi, cols db*16 + r) ------------------------------
#pragma unroll
  for (int db = 0; db < 4; ++db)
#pragma unroll
    for (int g = 0; g < 8; ++g)
      Ob[(size_t)(g + 8 * hi) * DD + db * 16 + r] = oacc[db][g];
}

extern "C" void kernel_launch(void* const* d_in, const int* in_sizes, int n_in,
                              void* d_out, int out_size, void* d_ws, size_t ws_size,
                              hipStream_t stream) {
  (void)in_sizes; (void)n_in; (void)out_size; (void)d_ws; (void)ws_size;
  const float* Q = (const float*)d_in[0];
  const float* K = (const float*)d_in[1];
  const float* V = (const float*)d_in[2];
  float* Out = (float*)d_out;                                // [B,S,D]
  float* Wt  = Out + (size_t)BB * SS * DD;                   // [B,S,S]
  // 2048 waves = B * S/16 query tiles; 8 waves (256 thr) per block
  fa_causal_fwd<<<dim3((BB * (SS / 16)) / 8), dim3(256), 0, stream>>>(Q, K, V, Out, Wt);
}
